// DigitCaps_80487687127186
// MI455X (gfx1250) — matrix operand order repaired and checked by reading, not compiled, not run on hardware
//
#include <hip/hip_runtime.h>

#define N_PRIMARY 1152
#define N_DIGIT   10
#define D_OUT     16
#define D_IN      8
#define BATCH     512

typedef __attribute__((ext_vector_type(16))) _Float16 v16h;
typedef __attribute__((ext_vector_type(8)))  _Float16 v8h;
typedef __attribute__((ext_vector_type(8)))  float    v8f;

// ---------------- prep ----------------
// One thread per (b,n): emit f16 copy of u in both [b,n,j] and transposed [n,j,b] layouts.
__global__ void k_prep_u(const float* __restrict__ u, _Float16* __restrict__ uh,
                         _Float16* __restrict__ uh_t) {
    int i = blockIdx.x * blockDim.x + threadIdx.x;       // over BATCH*N_PRIMARY
    if (i >= BATCH * N_PRIMARY) return;
    int b = i / N_PRIMARY, n = i % N_PRIMARY;
    const float* src = u + (size_t)i * D_IN;
    _Float16*    dst = uh + (size_t)i * D_IN;
    #pragma unroll
    for (int j = 0; j < D_IN; ++j) {
        _Float16 hv = (_Float16)src[j];
        dst[j] = hv;
        uh_t[((size_t)n * D_IN + j) * BATCH + b] = hv;
    }
}

__global__ void k_f32_to_f16(const float* __restrict__ src, _Float16* __restrict__ dst, int n) {
    int i = blockIdx.x * blockDim.x + threadIdx.x;
    if (i < n) dst[i] = (_Float16)src[i];
}

__global__ void k_zero(float* __restrict__ p, int n) {
    int i = blockIdx.x * blockDim.x + threadIdx.x;
    if (i < n) p[i] = 0.0f;
}

// softmax over the 10 digit capsules; emit routing coefficients directly in f16
__global__ void k_softmax(const float* __restrict__ blog, _Float16* __restrict__ cc) {
    int n = blockIdx.x * blockDim.x + threadIdx.x;
    if (n >= N_PRIMARY) return;
    float x[N_DIGIT];
    float m = -1e30f;
    #pragma unroll
    for (int c = 0; c < N_DIGIT; ++c) { x[c] = blog[n * N_DIGIT + c]; m = fmaxf(m, x[c]); }
    float sum = 0.0f;
    #pragma unroll
    for (int c = 0; c < N_DIGIT; ++c) { x[c] = __expf(x[c] - m); sum += x[c]; }
    float inv = 1.0f / sum;
    #pragma unroll
    for (int c = 0; c < N_DIGIT; ++c) cc[n * N_DIGIT + c] = (_Float16)(x[c] * inv);
}

// v = (|s|/(1+|s|^2)) * s.  For routing iterations, emit v transposed in f16:
// v_t[(c*16+i)*BATCH + b], which is the B-matrix layout for the b-update GEMM.
// On the final iteration, emit f32 [b,c,i] to d_out.
__global__ void k_squash(const float* __restrict__ s, float* __restrict__ out_f32,
                         _Float16* __restrict__ v_t, int final_iter) {
    int idx = blockIdx.x * blockDim.x + threadIdx.x;     // over BATCH*N_DIGIT
    if (idx >= BATCH * N_DIGIT) return;
    int b = idx / N_DIGIT, c = idx % N_DIGIT;
    const float* sp = s + (size_t)idx * D_OUT;
    float x[D_OUT];
    float n2 = 0.0f;
    #pragma unroll
    for (int i = 0; i < D_OUT; ++i) { x[i] = sp[i]; n2 += x[i] * x[i]; }
    float factor = sqrtf(n2) / (1.0f + n2);
    if (final_iter) {
        float* vp = out_f32 + (size_t)idx * D_OUT;
        #pragma unroll
        for (int i = 0; i < D_OUT; ++i) vp[i] = factor * x[i];
    } else {
        #pragma unroll
        for (int i = 0; i < D_OUT; ++i)
            v_t[((size_t)(c * D_OUT + i)) * BATCH + b] = (_Float16)(factor * x[i]);
    }
}

// ---------------- s = einsum('bnci,nc->bci') with u_hat recomputed in WMMA ----------------
// One wave per (16-batch tile, 32-n chunk). K=32 packs FOUR n's (K=8 each); the routing
// coefficient cc[n,c] is folded into B so the WMMA accumulator performs the n-reduction.
__global__ __launch_bounds__(32) void k_s_wmma(const _Float16* __restrict__ uh,
                                               const _Float16* __restrict__ Wh,
                                               const _Float16* __restrict__ cc,
                                               float* __restrict__ s) {
    const int b0    = blockIdx.x * 16;       // batch tile
    const int nbase = blockIdx.y * 32;       // n chunk
    const int lane  = threadIdx.x;
    const int h     = lane >> 4;             // lane half-group
    const int l16   = lane & 15;

    v8f acc[N_DIGIT];
    #pragma unroll
    for (int c = 0; c < N_DIGIT; ++c) { v8f z = {}; acc[c] = z; }

    for (int g = 0; g < 8; ++g) {            // 8 groups of 4 n's
        const int ng = nbase + 4 * g;
        // A (16x32 f16): row M = l16 = batch; lanes 0-15 hold K0-7 (n=ng+0) / K16-23 (n=ng+2),
        // lanes 16-31 hold K8-15 (n=ng+1) / K24-31 (n=ng+3).
        const int brow = b0 + l16;
        union { v16h v; v8h half8[2]; } A;
        A.half8[0] = *(const v8h*)(uh + ((size_t)brow * N_PRIMARY + (ng + h)) * D_IN);
        A.half8[1] = *(const v8h*)(uh + ((size_t)brow * N_PRIMARY + (ng + h + 2)) * D_IN);

        const int n_lo = ng + h;             // B: lanes0-15 K0-7/K16-23 -> q=0,2 ; lanes16-31 K8-15/K24-31 -> q=1,3
        const int n_hi = n_lo + 2;
        #pragma unroll
        for (int c = 0; c < N_DIGIT; ++c) {
            const _Float16 c_lo = cc[n_lo * N_DIGIT + c];
            const _Float16 c_hi = cc[n_hi * N_DIGIT + c];
            v8h wlo = *(const v8h*)(Wh + (((size_t)n_lo * N_DIGIT + c) * D_OUT + l16) * D_IN);
            v8h whi = *(const v8h*)(Wh + (((size_t)n_hi * N_DIGIT + c) * D_OUT + l16) * D_IN);
            union { v16h v; _Float16 e[16]; } B;
            #pragma unroll
            for (int t = 0; t < 8; ++t) { B.e[t] = wlo[t] * c_lo; B.e[8 + t] = whi[t] * c_hi; }
            acc[c] = __builtin_amdgcn_wmma_f32_16x16x32_f16(
                false, A.v, false, B.v, (short)0, acc[c], false, false);
        }
    }

    // D layout: VGPR k -> row k + 8*h, col = l16. Accumulate partial s across n-chunks.
    #pragma unroll
    for (int c = 0; c < N_DIGIT; ++c) {
        #pragma unroll
        for (int k = 0; k < 8; ++k) {
            atomicAdd(&s[(size_t)(b0 + k + 8 * h) * (N_DIGIT * D_OUT) + c * D_OUT + l16],
                      acc[c][k]);
        }
    }
}

// ---------------- b[n,c] += sum_{b,i} u_hat[b,n,c,i] * v[b,c,i] ----------------
// Full-K restructure: one wave per n computes, for all 10 c,
//   D[j,i] = sum_b u[b,n,j] * v[b,c,i]      (GEMM, M=j(8 of 16), N=i(16), K=batch=512)
// as 16 fully-packed K=32 WMMA steps, then contracts with f32 W[n,c,i,j] in-register.
// One wave owns the whole batch reduction -> plain (non-atomic) logit update.
__global__ __launch_bounds__(32) void k_bupd_wmma(const _Float16* __restrict__ uh_t,
                                                  const _Float16* __restrict__ v_t,
                                                  const float* __restrict__ W,
                                                  float* __restrict__ blog) {
    const int n    = blockIdx.x;
    const int lane = threadIdx.x;
    const int h    = lane >> 4;
    const int l16  = lane & 15;

    v8f acc[N_DIGIT];
    #pragma unroll
    for (int c = 0; c < N_DIGIT; ++c) { v8f z = {}; acc[c] = z; }

    for (int step = 0; step < BATCH / 32; ++step) {
        const int kb = step * 32;
        // A (16x32): row M = j = l16 (only j<8 real); lanes h=0: K kb+0..15,
        // lanes h=1: K kb+16..31.
        union { v16h v; v8h half8[2]; } A;
        { v16h z = {}; A.v = z; }
        if (l16 < D_IN) {
            const _Float16* ap = uh_t + ((size_t)n * D_IN + l16) * BATCH + kb + 16 * h;
            A.half8[0] = *(const v8h*)(ap);
            A.half8[1] = *(const v8h*)(ap + 8);
        }
        #pragma unroll
        for (int c = 0; c < N_DIGIT; ++c) {
            // B (32x16): col N = i = l16; lanes 0-15 hold K kb+0..15, lanes 16-31 K kb+16..31.
            v16h B = *(const v16h*)(v_t + ((size_t)(c * D_OUT + l16)) * BATCH + kb + 16 * h);
            acc[c] = __builtin_amdgcn_wmma_f32_16x16x32_f16(
                false, A.v, false, B, (short)0, acc[c], false, false);
        }
    }

    // D layout: VGPR k -> row j = k + 8*h (rows >=8 are zero), col i = l16.
    // Contract with W[n,c,i,j] (8 contiguous f32 per lane), wave-reduce, update logits.
    #pragma unroll
    for (int c = 0; c < N_DIGIT; ++c) {
        float p = 0.0f;
        if (h == 0) {
            const float* wp = W + (((size_t)n * N_DIGIT + c) * D_OUT + l16) * D_IN;
            #pragma unroll
            for (int k = 0; k < 8; ++k) p += acc[c][k] * wp[k];
        }
        #pragma unroll
        for (int off = 16; off >= 1; off >>= 1) p += __shfl_xor(p, off, 32);
        if (lane == 0) blog[n * N_DIGIT + c] += p;   // unique owner: no atomic needed
    }
}

// ---------------- launcher ----------------
extern "C" void kernel_launch(void* const* d_in, const int* in_sizes, int n_in,
                              void* d_out, int out_size, void* d_ws, size_t ws_size,
                              hipStream_t stream) {
    (void)in_sizes; (void)n_in; (void)out_size; (void)ws_size;
    const float* u = (const float*)d_in[0];
    const float* W = (const float*)d_in[1];
    float* out = (float*)d_out;

    char*  ws  = (char*)d_ws;
    size_t off = 0;
    auto carve = [&](size_t bytes) -> void* {
        void* p = ws + off;
        off += (bytes + 255) & ~(size_t)255;
        return p;
    };
    const int NU = BATCH * N_PRIMARY * D_IN;                 // 4.7M
    const int NW = N_PRIMARY * N_DIGIT * D_OUT * D_IN;       // 1.47M
    const int NB = N_PRIMARY * N_DIGIT;                      // 11520
    const int NS = BATCH * N_DIGIT * D_OUT;                  // 81920

    _Float16* uh   = (_Float16*)carve((size_t)NU * sizeof(_Float16));
    _Float16* uh_t = (_Float16*)carve((size_t)NU * sizeof(_Float16));
    _Float16* Wh   = (_Float16*)carve((size_t)NW * sizeof(_Float16));
    float*    blog = (float*)carve((size_t)NB * sizeof(float));
    _Float16* cc   = (_Float16*)carve((size_t)NB * sizeof(_Float16));
    float*    s    = (float*)carve((size_t)NS * sizeof(float));
    _Float16* v_t  = (_Float16*)carve((size_t)NS * sizeof(_Float16));

    // one-time prep (deterministic every call)
    const int NBN = BATCH * N_PRIMARY;
    k_prep_u<<<(NBN + 255) / 256, 256, 0, stream>>>(u, uh, uh_t);
    k_f32_to_f16<<<(NW + 255) / 256, 256, 0, stream>>>(W, Wh, NW);
    k_zero<<<(NB + 255) / 256, 256, 0, stream>>>(blog, NB);

    for (int r = 0; r < 3; ++r) {
        k_softmax<<<(N_PRIMARY + 127) / 128, 128, 0, stream>>>(blog, cc);
        k_zero<<<(NS + 255) / 256, 256, 0, stream>>>(s, NS);
        k_s_wmma<<<dim3(BATCH / 16, N_PRIMARY / 32), 32, 0, stream>>>(uh, Wh, cc, s);
        k_squash<<<(BATCH * N_DIGIT + 127) / 128, 128, 0, stream>>>(
            s, out, v_t, (r == 2) ? 1 : 0);
        if (r < 2) {
            k_bupd_wmma<<<N_PRIMARY, 32, 0, stream>>>(uh_t, v_t, W, blog);
        }
    }
}
